// PlainSelfAttention_87411174408369
// MI455X (gfx1250) — compile-verified, matrix operand order checked by reading, not run on hardware
//
#include <hip/hip_runtime.h>

// ---------------------------------------------------------------------------
// 2quad attention for MI455X (gfx1250, wave32, WMMA + TDM).
//
//   1) convert x -> bf16; transpose-convert Wqkv, Wo -> bf16 [N][K]
//   2) GEMM qkv = x @ Wqkv (TDM-staged LDS tiles), epilogue scatters into
//      q[b,h,t,d], k[b,h,t,d], vT[b,h,d,t]
//   3) fused 2quad attention (flash-style, no score materialization;
//      rowsum normalization only -- no softmax max tracking needed)
//   4) GEMM out = attn @ Wo (fp32 store to d_out)
// ---------------------------------------------------------------------------

#define BB 2
#define TT 2048
#define CC 1024
#define NH 16
#define DD 64

typedef __attribute__((ext_vector_type(16))) __bf16 v16bf;
typedef __attribute__((ext_vector_type(8)))  float  v8f;
typedef __attribute__((ext_vector_type(4)))  unsigned int v4u;
typedef __attribute__((ext_vector_type(8)))  int v8i;
typedef __attribute__((ext_vector_type(4)))  int v4i;

struct Frag32B { uint4 a, b; };

// A-layout (and B^T-layout) 16x32 bf16 fragment load from a row-major
// [16 rows][stride] buffer.  ISA 16-bit A layout:
//   lanes 0-15: m=lane,    VGPR0-3 K=0..7,  VGPR4-7 K=16..23
//   lanes16-31: m=lane-16, VGPR0-3 K=8..15, VGPR4-7 K=24..31
__device__ __forceinline__ v16bf load_fragA(const __bf16* base, unsigned stride,
                                            int lane) {
    const unsigned half = (unsigned)lane >> 4;
    const unsigned m    = (unsigned)lane & 15u;
    const __bf16* p = base + m * stride + half * 8u;
    Frag32B t;
    t.a = *reinterpret_cast<const uint4*>(p);        // K = half*8 .. +7
    t.b = *reinterpret_cast<const uint4*>(p + 16);   // K = 16+half*8 .. +7
    return __builtin_bit_cast(v16bf, t);
}

__device__ __forceinline__ v8f wmma_bf16(v16bf a, v16bf b, v8f c) {
    return __builtin_amdgcn_wmma_f32_16x16x32_bf16(false, a, false, b, (short)0, c,
                                                   false, false);
}

__device__ __forceinline__ v8f v8f_zero() {
    v8f z = {0.f, 0.f, 0.f, 0.f, 0.f, 0.f, 0.f, 0.f};
    return z;
}

// Wave-local LDS ordering (staging buffer is wave-private; same-wave DS ops
// are in-order in HW -- we only need to stop compiler reordering).
__device__ __forceinline__ void wave_lds_fence() {
    __builtin_amdgcn_fence(__ATOMIC_RELEASE, "wavefront");
    __builtin_amdgcn_wave_barrier();
    __builtin_amdgcn_fence(__ATOMIC_ACQUIRE, "wavefront");
}

// ---------------------------------------------------------------------------
// TDM: DMA a 128-row x 32-element bf16 tile (rows strided by rowStride elems)
// from global memory into LDS, padding 16 bytes after every 64-byte row so the
// LDS image has stride 40 bf16 (bank-conflict-free fragment loads).
// D# layout per CDNA5 ISA ch.8 (group0: count/lds_addr/global_addr/type;
// group1: data_size, pad cfg, tensor dims, tile dims, dim0 stride).
// This toolchain's builtin is the 6-arg form:
//   (uint32x4 g0, int32x8 g1, int32x4 g2, int32x4 g3, int32x8 gx, int cpol)
// ---------------------------------------------------------------------------
__device__ __forceinline__ void tdm_load_tile_128x32(const __bf16* src,
                                                     unsigned rowStride,
                                                     unsigned nRows,
                                                     unsigned ldsAddr) {
    const unsigned long long ga = (unsigned long long)(uintptr_t)src;
    v4u g0;
    g0[0] = 1u;                                       // count=1, user descriptor
    g0[1] = ldsAddr;                                  // lds_addr [63:32]
    g0[2] = (unsigned)ga;                             // global_addr [95:64]
    g0[3] = (unsigned)((ga >> 32) & 0x01FFFFFFu)      // global_addr [120:96]
          | (2u << 30);                               // type=2 (image) [127:126]
    v8i g1;
    g1[0] = (int)((1u << 16)      // data_size = 2 bytes (bf16)
                | (1u << 20)      // pad_enable
                | (3u << 22)      // pad_interval: 16 DWORDs (=64B row)
                | (3u << 25));    // pad_amount:   4 DWORDs (=16B pad)
    g1[1] = (int)((rowStride & 0xFFFFu) << 16);               // tensor_dim0 lo16
    g1[2] = (int)((rowStride >> 16) | ((nRows & 0xFFFFu) << 16)); // dim0 hi / dim1 lo
    g1[3] = (int)((nRows >> 16) | (32u << 16));               // dim1 hi, tile_dim0=32
    g1[4] = (int)128;                                         // tile_dim1=128
    g1[5] = (int)rowStride;                                   // tensor_dim0_stride
    g1[6] = 0;
    g1[7] = 0;
    v4i gz = {0, 0, 0, 0};
    v8i gz8 = {0, 0, 0, 0, 0, 0, 0, 0};
    __builtin_amdgcn_tensor_load_to_lds(g0, g1, gz, gz, gz8, 0);
}

// ---------------------------------------------------------------------------
// fp32 -> bf16 elementwise convert (n divisible by 1024)
// ---------------------------------------------------------------------------
__global__ void __launch_bounds__(256)
convert_bf16_kernel(const float* __restrict__ in, __bf16* __restrict__ out, int n) {
    unsigned i = (blockIdx.x * 256u + threadIdx.x) * 4u;
    if ((int)i + 3 < n) {
        out[i + 0] = (__bf16)in[i + 0];
        out[i + 1] = (__bf16)in[i + 1];
        out[i + 2] = (__bf16)in[i + 2];
        out[i + 3] = (__bf16)in[i + 3];
    }
}

// ---------------------------------------------------------------------------
// W [K][N] fp32  ->  Wt [N][K] bf16   (K, N multiples of 32)
// ---------------------------------------------------------------------------
__global__ void __launch_bounds__(256)
transpose_convert_kernel(const float* __restrict__ in, __bf16* __restrict__ out,
                         int K, int N) {
    __shared__ float tile[32][33];
    const unsigned kBase = blockIdx.y * 32u;
    const unsigned nBase = blockIdx.x * 32u;
    for (unsigned r = threadIdx.y; r < 32; r += 8)
        tile[r][threadIdx.x] = in[(kBase + r) * (unsigned)N + nBase + threadIdx.x];
    __syncthreads();
    for (unsigned r = threadIdx.y; r < 32; r += 8)
        out[(nBase + r) * (unsigned)K + kBase + threadIdx.x] =
            (__bf16)tile[threadIdx.x][r];
}

// ---------------------------------------------------------------------------
// bf16 GEMM:  C[M][N] = A[M][K] @ Bt[N][K]^T, fp32 accumulate.
// Block: 256 threads = 8 waves, tile 128x128, K-step 32; tiles staged to LDS
// by the Tensor Data Mover (wave 0 issues, TENSORcnt-tracked).
// Wave grid 4(m) x 2(n): each wave owns 32x64 = 2x4 WMMA tiles.
// MODE 0: scatter qkv epilogue (q, k row-major over d; v transposed)
// MODE 1: plain fp32 store
// M, N multiples of 128; K multiple of 32; all offsets < 2^31 -> 32-bit math.
// ---------------------------------------------------------------------------
template <int MODE>
__global__ void __launch_bounds__(256)
gemm_bf16_wmma(const __bf16* __restrict__ A, const __bf16* __restrict__ Bt,
               int M, int N, int K,
               __bf16* __restrict__ qb, __bf16* __restrict__ kb,
               __bf16* __restrict__ vtb, float* __restrict__ outF) {
    __shared__ __align__(16) __bf16 sA[128 * 40];   // 128 rows, 32 K, pad->40
    __shared__ __align__(16) __bf16 sB[128 * 40];

    const int tid  = threadIdx.x;
    const int lane = tid & 31;
    const int wave = tid >> 5;
    const int wm   = wave & 3;        // 0..3 -> 32-row slab
    const int wn   = wave >> 2;       // 0..1 -> 64-col slab
    const unsigned m0 = blockIdx.y * 128u;
    const unsigned n0 = blockIdx.x * 128u;
    const unsigned uK = (unsigned)K;

    v8f acc[2][4];
    for (int mt = 0; mt < 2; ++mt)
        for (int nt = 0; nt < 4; ++nt)
            acc[mt][nt] = v8f_zero();

    const unsigned ldsA = (unsigned)(uintptr_t)(void*)sA;
    const unsigned ldsB = (unsigned)(uintptr_t)(void*)sB;

    for (unsigned k0 = 0; k0 < uK; k0 += 32u) {
        if (wave == 0) {
            tdm_load_tile_128x32(A + m0 * uK + k0, uK, (unsigned)M, ldsA);
            tdm_load_tile_128x32(Bt + n0 * uK + k0, uK, (unsigned)N, ldsB);
            __builtin_amdgcn_s_wait_tensorcnt(0);
        }
        __syncthreads();

        v16bf af[2], bfr[4];
        for (int mt = 0; mt < 2; ++mt)
            af[mt] = load_fragA(&sA[(wm * 32 + mt * 16) * 40], 40u, lane);
        for (int nt = 0; nt < 4; ++nt)
            bfr[nt] = load_fragA(&sB[(wn * 64 + nt * 16) * 40], 40u, lane);

        for (int mt = 0; mt < 2; ++mt)
            for (int nt = 0; nt < 4; ++nt)
                acc[mt][nt] = wmma_bf16(af[mt], bfr[nt], acc[mt][nt]);
        __syncthreads();   // protect LDS from next iteration's TDM overwrite
    }

    // Epilogue. C layout: vgpr v -> row m = v + 8*(lane/16), col n = lane%16.
    const unsigned half = (unsigned)lane >> 4;
    const unsigned colv = (unsigned)lane & 15u;
    for (int mt = 0; mt < 2; ++mt) {
        for (int nt = 0; nt < 4; ++nt) {
            for (int v = 0; v < 8; ++v) {
                const unsigned gm = m0 + wm * 32u + mt * 16u + v + 8u * half;
                const unsigned gn = n0 + wn * 64u + nt * 16u + colv;
                const float val = acc[mt][nt][v];
                if (MODE == 0) {
                    const unsigned region = gn >> 10;       // 0=q 1=k 2=v
                    const unsigned c  = gn & 1023u;
                    const unsigned hh = c >> 6;
                    const unsigned dd = c & 63u;
                    const unsigned bb = gm >> 11;
                    const unsigned t  = gm & 2047u;
                    const unsigned bh = bb * NH + hh;
                    if (region == 0)
                        qb[(bh * TT + t) * DD + dd] = (__bf16)val;
                    else if (region == 1)
                        kb[(bh * TT + t) * DD + dd] = (__bf16)val;
                    else
                        vtb[(bh * DD + dd) * TT + t] = (__bf16)val;
                } else {
                    outF[gm * (unsigned)N + gn] = val;
                }
            }
        }
    }
}

// ---------------------------------------------------------------------------
// Fused 2quad attention.
// grid = (T/128, B*NH), block = 256 (8 waves). Wave w owns q rows
// [bx*128 + w*16, +16) -- fully independent waves (wave-private LDS staging).
// Loop over 64 chunks of 32 keys:
//   s(16x16) x2 via WMMA over d=64; e=(s*scale+5)^2; rowsum += e;
//   e -> per-wave LDS (C->A layout fixup); out(16x64) += e @ v via 4 WMMA.
// Normalize by (rowsum + eps), store bf16 at [b,t,h*64+dd] for the Wo GEMM.
// ---------------------------------------------------------------------------
__global__ void __launch_bounds__(256)
attn2quad_kernel(const __bf16* __restrict__ q, const __bf16* __restrict__ k,
                 const __bf16* __restrict__ vT, const float* __restrict__ tau,
                 __bf16* __restrict__ attn) {
    __shared__ __align__(16) __bf16 se[8 * 16 * 40];   // per-wave 16x32 e tile

    const int tid  = threadIdx.x;
    const int lane = tid & 31;
    const int wave = tid >> 5;
    const unsigned half = (unsigned)lane >> 4;
    const unsigned col  = (unsigned)lane & 15u;
    const unsigned bh   = blockIdx.y;
    const unsigned b    = bh >> 4;
    const unsigned h    = bh & 15u;
    const unsigned q0   = blockIdx.x * 128u + wave * 16u;

    const __bf16* qh = q  + bh * (TT * DD);
    const __bf16* kh = k  + bh * (TT * DD);
    const __bf16* vh = vT + bh * (DD * TT);
    const float scale = 1.0f / (8.0f * tau[h]);   // sqrt(64)=8

    // q fragments: rows contiguous over d -> direct loads (d 0..31 / 32..63)
    const v16bf qf0 = load_fragA(qh + q0 * DD,       DD, lane);
    const v16bf qf1 = load_fragA(qh + q0 * DD + 32u, DD, lane);

    v8f o[4];
    for (int nt = 0; nt < 4; ++nt) o[nt] = v8f_zero();
    float rs[8] = {};
    __bf16* sew = se + wave * (16 * 40);

    for (unsigned kt = 0; kt < TT / 32; ++kt) {
        // prefetch next chunk's k rows and v rows (global_prefetch_b8)
        if (kt + 1 < TT / 32) {
            __builtin_prefetch(kh + (kt + 1) * 32u * DD + (unsigned)lane * 64u, 0, 1);
            __builtin_prefetch(vh + (unsigned)lane * 2u * TT + (kt + 1) * 32u, 0, 1);
        }
        for (unsigned sub = 0; sub < 2; ++sub) {
            const unsigned kbase = kt * 32u + sub * 16u;
            const v16bf kf0 = load_fragA(kh + kbase * DD,       DD, lane);
            const v16bf kf1 = load_fragA(kh + kbase * DD + 32u, DD, lane);
            v8f s = v8f_zero();
            s = wmma_bf16(qf0, kf0, s);
            s = wmma_bf16(qf1, kf1, s);
            for (int v = 0; v < 8; ++v) {
                float e = s[v] * scale + 5.0f;   // 2quad: (s + c)^2
                e = e * e;
                rs[v] += e;                       // per-lane partial row sum
                sew[(v + 8u * half) * 40u + sub * 16u + col] = (__bf16)e;
            }
        }
        wave_lds_fence();   // wave-private staging: same-wave DS ops are in-order

        const v16bf ef = load_fragA(sew, 40u, lane);     // e as 16x32 A fragment
        for (unsigned nt = 0; nt < 4; ++nt) {
            const v16bf vf = load_fragA(vh + nt * 16u * TT + kt * 32u, TT, lane);
            o[nt] = wmma_bf16(ef, vf, o[nt]);
        }
        wave_lds_fence();   // WAR: next iteration overwrites the staging tile
    }

    // Cross-lane row-sum reduction within each 16-lane half, then normalize.
    for (int v = 0; v < 8; ++v) {
        float r = rs[v];
        r += __shfl_xor(r, 8, 32);
        r += __shfl_xor(r, 4, 32);
        r += __shfl_xor(r, 2, 32);
        r += __shfl_xor(r, 1, 32);
        const float inv = 1.0f / (r + 1e-6f);
        const unsigned t = q0 + (unsigned)v + 8u * half;
        const unsigned base = (b * TT + t) * CC + h * DD;
        for (unsigned nt = 0; nt < 4; ++nt)
            attn[base + nt * 16u + col] = (__bf16)(o[nt][v] * inv);
    }
}

// ---------------------------------------------------------------------------
// Launch
// ---------------------------------------------------------------------------
extern "C" void kernel_launch(void* const* d_in, const int* in_sizes, int n_in,
                              void* d_out, int out_size, void* d_ws, size_t ws_size,
                              hipStream_t stream) {
    (void)in_sizes; (void)n_in; (void)out_size; (void)ws_size;
    const float* x    = (const float*)d_in[0];
    const float* Wqkv = (const float*)d_in[1];
    const float* Wo   = (const float*)d_in[2];
    const float* tau  = (const float*)d_in[3];
    float* out = (float*)d_out;

    char* ws = (char*)d_ws;
    const int M = BB * TT;                                      // 4096
    __bf16* xb    = (__bf16*)(ws);                              //  8 MB
    __bf16* wqkvT = (__bf16*)(ws + 8  * 1024 * 1024);           //  6 MB
    __bf16* woT   = (__bf16*)(ws + 14 * 1024 * 1024);           //  2 MB
    __bf16* qbuf  = (__bf16*)(ws + 16 * 1024 * 1024);           //  8 MB
    __bf16* kbuf  = (__bf16*)(ws + 24 * 1024 * 1024);           //  8 MB
    __bf16* vtbuf = (__bf16*)(ws + 32 * 1024 * 1024);           //  8 MB
    __bf16* attnb = (__bf16*)(ws + 40 * 1024 * 1024);           //  8 MB (48 MB)

    // 1) converts
    {
        const int n = M * CC;
        convert_bf16_kernel<<<n / (256 * 4), 256, 0, stream>>>(x, xb, n);
    }
    transpose_convert_kernel<<<dim3(3 * CC / 32, CC / 32), dim3(32, 8), 0, stream>>>(
        Wqkv, wqkvT, CC, 3 * CC);
    transpose_convert_kernel<<<dim3(CC / 32, CC / 32), dim3(32, 8), 0, stream>>>(
        Wo, woT, CC, CC);

    // 2) qkv GEMM + head scatter (M=4096, N=3072, K=1024)
    gemm_bf16_wmma<0><<<dim3(3 * CC / 128, M / 128), 256, 0, stream>>>(
        xb, wqkvT, M, 3 * CC, CC, qbuf, kbuf, vtbuf, nullptr);

    // 3) fused 2quad attention
    attn2quad_kernel<<<dim3(TT / 128, BB * NH), 256, 0, stream>>>(
        qbuf, kbuf, vtbuf, tau, attnb);

    // 4) output projection (M=4096, N=1024, K=1024), fp32 store
    gemm_bf16_wmma<1><<<dim3(CC / 128, M / 128), 256, 0, stream>>>(
        attnb, woT, M, CC, CC, nullptr, nullptr, nullptr, out);
}
